// Attention_57543971832746
// MI455X (gfx1250) — matrix operand order repaired and checked by reading, not compiled, not run on hardware
//
#include <hip/hip_runtime.h>
#include <cstdint>

// ---------------------------------------------------------------- constants
#define BATCH   2
#define SEQ     2048
#define DIM     2048
#define NH      32
#define NKV     8
#define HD      64
#define QKV_E   3072            // (NH + 2*NKV) * HD
#define KV_DIM  512             // NKV * HD
#define MROWS   (BATCH * SEQ)   // 4096

typedef __attribute__((ext_vector_type(16))) __bf16 bf16x16;
typedef __attribute__((ext_vector_type(8)))  float  f32x8;

union FragB16 {
    bf16x16 v;
    uint4   x4[2];
    unsigned int   d[8];
    unsigned short u[16];
};

static __device__ __forceinline__ unsigned short f2bf_u(float f) {
    union { float f; unsigned int u; } c; c.f = f;
    unsigned int u = c.u;
    unsigned int r = u + 0x7FFFu + ((u >> 16) & 1u);   // RNE to bf16
    return (unsigned short)(r >> 16);
}

static __device__ __forceinline__ f32x8 wmma_bf16(const FragB16& a, const FragB16& b, f32x8 c) {
    return __builtin_amdgcn_wmma_f32_16x16x32_bf16(
        false, a.v, false, b.v, (short)0, c, false, false);
}

// ---------------------------------------------------------------- fp32 -> bf16
__global__ __launch_bounds__(256) void f32_to_bf16_kernel(
    const float* __restrict__ src, unsigned short* __restrict__ dst, int n)
{
    int i = (blockIdx.x * 256 + threadIdx.x) * 4;
    if (i + 3 < n) {
        float4 f = *(const float4*)(src + i);
        dst[i + 0] = f2bf_u(f.x);
        dst[i + 1] = f2bf_u(f.y);
        dst[i + 2] = f2bf_u(f.z);
        dst[i + 3] = f2bf_u(f.w);
    } else {
        for (int j = i; j < n; ++j) dst[j] = f2bf_u(src[j]);
    }
}

// ---------------------------------------------------------------- GEMM (NT)
// C[m,n] = sum_k A[m,k] * B[n,k]; A:[M,K] bf16, B:[N,K] bf16, C:[M,N] f32
// block: 256 thr = 8 waves, wave tile 64x64, block tile 128x256
__global__ __launch_bounds__(256) void gemm_nt_bf16_kernel(
    const unsigned short* __restrict__ A,
    const unsigned short* __restrict__ Bm,
    float* __restrict__ C, int M, int N, int K)
{
    const int lane = threadIdx.x & 31;
    const int wave = threadIdx.x >> 5;
    const int hi   = lane >> 4;       // half of wave (0: lanes 0-15, 1: 16-31)
    const int l16  = lane & 15;
    const int wm = wave >> 2, wn = wave & 3;
    const int m0 = blockIdx.y * 128 + wm * 64;
    const int n0 = blockIdx.x * 256 + wn * 64;

    f32x8 acc[4][4];
#pragma unroll
    for (int i = 0; i < 4; ++i)
#pragma unroll
        for (int j = 0; j < 4; ++j)
            acc[i][j] = (f32x8)0.0f;

    for (int k0 = 0; k0 < K; k0 += 32) {
        FragB16 a[4], b[4];
#pragma unroll
        for (int i = 0; i < 4; ++i) {
            const unsigned short* ap =
                A + (size_t)(m0 + i * 16 + l16) * K + k0 + hi * 8;
            a[i].x4[0] = *(const uint4*)ap;         // K = k0+hi*8   .. +7
            a[i].x4[1] = *(const uint4*)(ap + 16);  // K = k0+16+hi*8.. +7
        }
#pragma unroll
        for (int j = 0; j < 4; ++j) {
            const unsigned short* bp =
                Bm + (size_t)(n0 + j * 16 + l16) * K + k0 + hi * 8;
            b[j].x4[0] = *(const uint4*)bp;         // K = k0+hi*8   .. +7
            b[j].x4[1] = *(const uint4*)(bp + 16);
        }
#pragma unroll
        for (int i = 0; i < 4; ++i)
#pragma unroll
            for (int j = 0; j < 4; ++j)
                acc[i][j] = wmma_bf16(a[i], b[j], acc[i][j]);
    }

#pragma unroll
    for (int i = 0; i < 4; ++i)
#pragma unroll
        for (int j = 0; j < 4; ++j)
#pragma unroll
            for (int r = 0; r < 8; ++r)
                C[(size_t)(m0 + i * 16 + hi * 8 + r) * N + (n0 + j * 16 + l16)]
                    = acc[i][j][r];
}

// ---------------------------------------------------------------- RoPE + pack
// q  -> [B*NH , S, 64] bf16 (rotated)
// k  -> [B*NKV, S, 64] bf16 (rotated)
// vT -> [B*NKV, 64, S] bf16 (transposed for P*V B-fragments)
#define Q_PAIRS (BATCH * SEQ * NH  * (HD / 2))   // 4194304
#define K_PAIRS (BATCH * SEQ * NKV * (HD / 2))   // 1048576
#define V_ELEMS (BATCH * SEQ * KV_DIM)           // 2097152

__global__ __launch_bounds__(256) void rope_pack_kernel(
    const float* __restrict__ qkv, const float* __restrict__ freqs,
    unsigned short* __restrict__ q, unsigned short* __restrict__ k,
    unsigned short* __restrict__ vT)
{
    int tid = blockIdx.x * 256 + threadIdx.x;
    if (tid < Q_PAIRS) {
        int p = tid & 31;
        int h = (tid >> 5) & (NH - 1);
        int s = (tid >> 10) & (SEQ - 1);
        int b = tid >> 21;
        size_t base = ((size_t)(b * SEQ + s)) * QKV_E + h * HD + 2 * p;
        float x0 = qkv[base], x1 = qkv[base + 1];
        float c = freqs[(s * 32 + p) * 2], sn = freqs[(s * 32 + p) * 2 + 1];
        size_t o = (((size_t)(b * NH + h)) * SEQ + s) * HD + 2 * p;
        q[o]     = f2bf_u(x0 * c - x1 * sn);
        q[o + 1] = f2bf_u(x1 * c + x0 * sn);
    } else if (tid < Q_PAIRS + K_PAIRS) {
        int t = tid - Q_PAIRS;
        int p = t & 31;
        int h = (t >> 5) & (NKV - 1);
        int s = (t >> 8) & (SEQ - 1);
        int b = t >> 19;
        size_t base = ((size_t)(b * SEQ + s)) * QKV_E + DIM + h * HD + 2 * p;
        float x0 = qkv[base], x1 = qkv[base + 1];
        float c = freqs[(s * 32 + p) * 2], sn = freqs[(s * 32 + p) * 2 + 1];
        size_t o = (((size_t)(b * NKV + h)) * SEQ + s) * HD + 2 * p;
        k[o]     = f2bf_u(x0 * c - x1 * sn);
        k[o + 1] = f2bf_u(x1 * c + x0 * sn);
    } else if (tid < Q_PAIRS + K_PAIRS + V_ELEMS) {
        int t = tid - Q_PAIRS - K_PAIRS;
        int d = t & (HD - 1);
        int h = (t >> 6) & (NKV - 1);
        int s = (t >> 9) & (SEQ - 1);
        int b = t >> 20;
        float v = qkv[((size_t)(b * SEQ + s)) * QKV_E + DIM + KV_DIM + h * HD + d];
        vT[(((size_t)(b * NKV + h)) * HD + d) * SEQ + s] = f2bf_u(v);
    }
}

// ---------------------------------------------------------------- flash attn
// block: 128 thr = 4 waves; block handles (b, h, 64 q-rows); each wave 16 rows
__global__ __launch_bounds__(128) void flash_attn_kernel(
    const unsigned short* __restrict__ q, const unsigned short* __restrict__ k,
    const unsigned short* __restrict__ vT, unsigned short* __restrict__ o)
{
    const int lane = threadIdx.x & 31;
    const int wave = threadIdx.x >> 5;
    const int hi   = lane >> 4;
    const int l16  = lane & 15;
    const int qb = blockIdx.x, h = blockIdx.y, b = blockIdx.z;
    const int hkv = h >> 2;                       // GQA: 4 q-heads per kv-head

    const unsigned short* qbase =
        q + (((size_t)(b * NH + h)) * SEQ + qb * 64 + wave * 16) * HD;
    const unsigned short* kbase = k  + ((size_t)(b * NKV + hkv)) * SEQ * HD;
    const unsigned short* vbase = vT + ((size_t)(b * NKV + hkv)) * HD * SEQ;

    __shared__ unsigned short p_lds[4][16][80];   // stride 80 keeps 16B align

    // Q fragments (A-matrix, 16x32 bf16), loaded once: hd=64 -> 2 chunks
    FragB16 aq[2];
#pragma unroll
    for (int kk = 0; kk < 2; ++kk) {
        const unsigned short* ap = qbase + (size_t)l16 * HD + kk * 32 + hi * 8;
        aq[kk].x4[0] = *(const uint4*)ap;
        aq[kk].x4[1] = *(const uint4*)(ap + 16);
    }

    f32x8 acco[4];
#pragma unroll
    for (int nt = 0; nt < 4; ++nt) acco[nt] = (f32x8)0.0f;
    float m_i[8], l_i[8];
#pragma unroll
    for (int r = 0; r < 8; ++r) { m_i[r] = -1.0e30f; l_i[r] = 0.0f; }

    const float scale = 0.125f;                   // 1/sqrt(64)
    const int qrow0 = qb * 64 + wave * 16 + hi * 8;

    for (int kb = 0; kb <= qb; ++kb) {
        const int kbeg = kb * 64;

        // ---- scores S = Q K^T (16 q-rows x 64 k-cols)
        f32x8 s[4];
#pragma unroll
        for (int nt = 0; nt < 4; ++nt) s[nt] = (f32x8)0.0f;
#pragma unroll
        for (int nt = 0; nt < 4; ++nt) {
#pragma unroll
            for (int kk = 0; kk < 2; ++kk) {
                FragB16 bk;
                const unsigned short* bp =
                    kbase + (size_t)(kbeg + nt * 16 + l16) * HD + kk * 32 + hi * 8;
                bk.x4[0] = *(const uint4*)bp;
                bk.x4[1] = *(const uint4*)(bp + 16);
                s[nt] = wmma_bf16(aq[kk], bk, s[nt]);
            }
        }

        // ---- causal mask on diagonal block
        if (kb == qb) {
#pragma unroll
            for (int nt = 0; nt < 4; ++nt)
#pragma unroll
                for (int r = 0; r < 8; ++r)
                    if (kbeg + nt * 16 + l16 > qrow0 + r) s[nt][r] = -1.0e30f;
        }

        // ---- online softmax: row max (reduce across the 16-lane half)
        float mnew[8];
#pragma unroll
        for (int r = 0; r < 8; ++r) {
            float v = fmaxf(fmaxf(s[0][r], s[1][r]), fmaxf(s[2][r], s[3][r]));
#pragma unroll
            for (int msk = 1; msk < 16; msk <<= 1)
                v = fmaxf(v, __shfl_xor(v, msk, 32));
            mnew[r] = fmaxf(m_i[r], v * scale);
        }
        float rowsum[8];
#pragma unroll
        for (int r = 0; r < 8; ++r) rowsum[r] = 0.0f;
#pragma unroll
        for (int nt = 0; nt < 4; ++nt)
#pragma unroll
            for (int r = 0; r < 8; ++r) {
                float e = __expf(s[nt][r] * scale - mnew[r]);
                s[nt][r] = e;
                rowsum[r] += e;
            }
#pragma unroll
        for (int r = 0; r < 8; ++r) {
            float v = rowsum[r];
#pragma unroll
            for (int msk = 1; msk < 16; msk <<= 1)
                v += __shfl_xor(v, msk, 32);
            float alpha = __expf(m_i[r] - mnew[r]);
            l_i[r] = l_i[r] * alpha + v;
            m_i[r] = mnew[r];
#pragma unroll
            for (int nt = 0; nt < 4; ++nt) acco[nt][r] *= alpha;
        }

        // ---- P through LDS: C-layout -> A-fragment layout
        __syncthreads();
#pragma unroll
        for (int nt = 0; nt < 4; ++nt)
#pragma unroll
            for (int r = 0; r < 8; ++r)
                p_lds[wave][hi * 8 + r][nt * 16 + l16] = f2bf_u(s[nt][r]);
        __syncthreads();

        // ---- O += P V  (contraction over 64 k-cols -> 2 chunks of 32)
#pragma unroll
        for (int kk = 0; kk < 2; ++kk) {
            FragB16 ap;
            const unsigned short* pp = &p_lds[wave][l16][kk * 32 + hi * 8];
            ap.x4[0] = *(const uint4*)pp;
            ap.x4[1] = *(const uint4*)(pp + 16);
#pragma unroll
            for (int nt = 0; nt < 4; ++nt) {
                FragB16 bv;
                const unsigned short* vp =
                    vbase + (size_t)(nt * 16 + l16) * SEQ + kbeg + kk * 32 + hi * 8;
                bv.x4[0] = *(const uint4*)vp;
                bv.x4[1] = *(const uint4*)(vp + 16);
                acco[nt] = wmma_bf16(ap, bv, acco[nt]);
            }
        }
        __syncthreads();
    }

    // ---- epilogue: O /= l, write bf16 [B, S, DIM]
#pragma unroll
    for (int nt = 0; nt < 4; ++nt)
#pragma unroll
        for (int r = 0; r < 8; ++r) {
            int qrow = qrow0 + r;
            int col  = h * HD + nt * 16 + l16;
            o[((size_t)b * SEQ + qrow) * DIM + col] = f2bf_u(acco[nt][r] / l_i[r]);
        }
}

// ---------------------------------------------------------------- launcher
extern "C" void kernel_launch(void* const* d_in, const int* in_sizes, int n_in,
                              void* d_out, int out_size, void* d_ws, size_t ws_size,
                              hipStream_t stream) {
    const float* x     = (const float*)d_in[0];   // [B,S,DIM]
    const float* freqs = (const float*)d_in[1];   // [S,32,2]
    const float* wqkv  = (const float*)d_in[2];   // [3072,2048]
    const float* wo    = (const float*)d_in[3];   // [2048,2048]
    float* out = (float*)d_out;

    char* w = (char*)d_ws;
    unsigned short* x_bf    = (unsigned short*)(w + 0);           // 16 MB
    unsigned short* wqkv_bf = (unsigned short*)(w + 16777216);    // 12 MB
    unsigned short* wo_bf   = (unsigned short*)(w + 29360128);    //  8 MB
    float*          qkv_f   = (float*)        (w + 37748736);    // 48 MB
    unsigned short* q_bf    = (unsigned short*)(w + 88080384);    // 16 MB
    unsigned short* k_bf    = (unsigned short*)(w + 104857600);   //  4 MB
    unsigned short* vT_bf   = (unsigned short*)(w + 109051904);   //  4 MB
    unsigned short* att_bf  = (unsigned short*)(w + 113246208);   // 16 MB

    // 1) fp32 -> bf16 staging
    f32_to_bf16_kernel<<<(MROWS * DIM + 1023) / 1024, 256, 0, stream>>>(x, x_bf, MROWS * DIM);
    f32_to_bf16_kernel<<<(QKV_E * DIM + 1023) / 1024, 256, 0, stream>>>(wqkv, wqkv_bf, QKV_E * DIM);
    f32_to_bf16_kernel<<<(DIM * DIM + 1023) / 1024, 256, 0, stream>>>(wo, wo_bf, DIM * DIM);

    // 2) QKV GEMM: [4096,2048] x [3072,2048]^T -> [4096,3072] f32
    gemm_nt_bf16_kernel<<<dim3(QKV_E / 256, MROWS / 128), 256, 0, stream>>>(
        x_bf, wqkv_bf, qkv_f, MROWS, QKV_E, DIM);

    // 3) RoPE + layout pack (q, k rotated; v transposed)
    rope_pack_kernel<<<(Q_PAIRS + K_PAIRS + V_ELEMS + 255) / 256, 256, 0, stream>>>(
        qkv_f, freqs, q_bf, k_bf, vT_bf);

    // 4) causal flash attention -> att_bf [B,S,DIM] bf16
    flash_attn_kernel<<<dim3(SEQ / 64, NH, BATCH), 128, 0, stream>>>(
        q_bf, k_bf, vT_bf, att_bf);

    // 5) output GEMM: [4096,2048] x [2048,2048]^T -> d_out f32
    gemm_nt_bf16_kernel<<<dim3(DIM / 256, MROWS / 128), 256, 0, stream>>>(
        att_bf, wo_bf, out, MROWS, DIM, DIM);
}
